// BasicQuadRGBV2Model_26233660244076
// MI455X (gfx1250) — matrix operand order repaired and checked by reading, not compiled, not run on hardware
//
#include <hip/hip_runtime.h>

// ---------------------------------------------------------------------------
// CDNA5 (gfx1250) fused demosaic: conv stacks via v_wmma_f32_16x16x32_f16.
// Kernel 1: both 3-layer 3x3 conv stacks + softmax + green, fully in LDS,
//           mosaic tile streamed in with GLOBAL_LOAD_ASYNC_TO_LDS (ASYNCcnt).
// Kernel 2: 5x5 chroma convs on the (sparse) pixel-shuffled plane + interleave.
// ---------------------------------------------------------------------------

typedef __attribute__((ext_vector_type(16))) _Float16 v16h;
typedef __attribute__((ext_vector_type(8)))  float    v8f;

#define TW   16          // output tile side (low-res pixels)
#define AS   26          // LDS activation array side (16 + halo + border pad)
#define CH   16          // padded channel count (WIDTH=12 -> 16)
#define KROW 160         // A-matrix row length: 10 taps * 16 channels
#define IMG  512
#define PLANE (512*512)
#define OPLANE (1024*1024)

#if defined(__has_builtin)
#if __has_builtin(__builtin_amdgcn_global_load_async_to_lds_b32)
#define ASYNC_LDS 1
#endif
#endif

union AFrag { v16h v; uint4 q[2]; };
union DFrag { v8f  v; float f[8]; };
union OPack { _Float16 h[8]; uint4 q; };

__device__ __forceinline__ const uint4* ldsq(const _Float16* p) {
  return reinterpret_cast<const uint4*>(p);
}

// One 3x3 conv layer (Cin<=16 -> 12 out ch, ReLU) over a 26x26x16 f16 LDS
// activation array, as implicit GEMM on WMMA 16x16x32 f16.
// K packing: k = tap*16 + c. A: lane M = lane%16, half h holds K = 32s+16h+e (e = 0..15).
// B: half h holds K = 32s+16h+e => one full tap channel vector.
// CENTER=false: compute 24x24 interior (36 groups). CENTER=true: only the
// final 16x16 output region (16 groups) -- used for both stacks' last layer.
template <bool CENTER>
__device__ __forceinline__
void conv3x3_wmma(const _Float16* __restrict__ src, _Float16* __restrict__ dst,
                  const _Float16* __restrict__ wA, const _Float16* __restrict__ zc)
{
  const int lane = threadIdx.x & 31;
  const int wave = threadIdx.x >> 5;     // 8 waves of wave32
  const int n    = lane & 15;            // N (pixel) / M (out channel) index
  const int h    = lane >> 4;            // lane half
  const int NG   = CENTER ? 16 : 36;
  for (int g = wave; g < NG; g += 8) {
    int ly, lx;
    if (CENTER) {                        // group g = output row g, col n
      ly = 4 + g;
      lx = 4 + n;
    } else {                             // 36 groups of 16 px = 24x24 region
      const int p  = g * 16 + n;         // 0..575
      const int py = p / 24;
      ly = 1 + py;
      lx = 1 + (p - py * 24);
    }
    DFrag acc;
    #pragma unroll
    for (int r = 0; r < 8; ++r) acc.f[r] = 0.0f;
    #pragma unroll
    for (int s = 0; s < 5; ++s) {        // K = 160 = 5 x 32
      AFrag a, bm;
      const _Float16* ar = wA + n * KROW + s * 32 + h * 16;
      a.q[0] = *ldsq(ar);                // tap 2s+h, c = 0..7
      a.q[1] = *ldsq(ar + 8);            // tap 2s+h, c = 8..15
      const int tb = 2 * s + h;          // B tap for this lane half
      const int ty = tb / 3;
      const int dy = ty - 1;
      const int dx = (tb - ty * 3) - 1;
      const _Float16* bp = (tb < 9)
          ? src + ((ly + dy) * AS + (lx + dx)) * CH
          : zc;                          // zero pad chunk (no EXEC divergence)
      bm.q[0] = *ldsq(bp);
      bm.q[1] = *ldsq(bp + 8);
      acc.v = __builtin_amdgcn_wmma_f32_16x16x32_f16(
          false, a.v, false, bm.v, (short)0, acc.v, false, false);
    }
    // D: lane holds out channels r+8h for pixel n; ReLU, pack, one b128 store.
    OPack o;
    #pragma unroll
    for (int r = 0; r < 8; ++r) o.h[r] = (_Float16)fmaxf(acc.f[r], 0.0f);
    *reinterpret_cast<uint4*>(dst + (ly * AS + lx) * CH + h * 8) = o.q;
  }
}

__global__ __launch_bounds__(256)
void demosaic_stacks_kernel(const float* __restrict__ mosaic,
                            const float* __restrict__ fw0, const float* __restrict__ fw1,
                            const float* __restrict__ fw2, const float* __restrict__ ww0,
                            const float* __restrict__ ww1, const float* __restrict__ ww2,
                            float* __restrict__ out, float* __restrict__ gws)
{
  __shared__ _Float16 S[3][AS * AS * CH];     // ping-pong activations (f16)
  __shared__ _Float16 wA[6][16 * KROW];       // per-layer A-matrix weights
  __shared__ _Float16 interpsF[TW * TW * CH]; // saved f-stack output
  __shared__ _Float16 zeroChunk[16];
  __shared__ float    mstage[22 * 22 * 4];    // raw f32 mosaic tile (async DMA)

  const int tid = threadIdx.x;
  const int b   = blockIdx.z;
  const int gx0 = blockIdx.x * TW;
  const int gy0 = blockIdx.y * TW;

#ifdef ASYNC_LDS
  // Kick off async global->LDS copies of the raw mosaic tile first, so the
  // DMA overlaps with the zeroing / weight staging VALU work below.
  for (int idx = tid; idx < 22 * 22 * 4; idx += 256) {
    const int c   = idx / 484;
    const int pos = idx - c * 484;
    const int j   = pos / 22;
    const int i   = pos - j * 22;
    const int iy  = gy0 + j - 3;
    const int ix  = gx0 + i - 3;
    if (iy >= 0 && iy < IMG && ix >= 0 && ix < IMG) {
      const float* gp = mosaic + ((size_t)b * 4 + c) * PLANE + (size_t)iy * IMG + ix;
      __builtin_amdgcn_global_load_async_to_lds_b32(
          (__attribute__((address_space(1))) int*)gp,
          (__attribute__((address_space(3))) int*)&mstage[idx], 0, 0);
    }
  }
#endif

  // Zero activation arrays (borders must stay zero => conv zero padding).
  {
    unsigned* z = reinterpret_cast<unsigned*>(&S[0][0]);
    const int nw = 3 * AS * AS * CH / 2;
    for (int i = tid; i < nw; i += 256) z[i] = 0u;
    if (tid < 16) zeroChunk[tid] = (_Float16)0.0f;
  }

  // Stage weights f32 -> f16 in A-matrix layout  wA[l][m][tap*16+c].
  {
    const float* Wg[6]  = {fw0, fw1, fw2, ww0, ww1, ww2};
    const int    Cin[6] = {4, 12, 12, 4, 12, 12};
    for (int idx = tid; idx < 6 * 16 * KROW; idx += 256) {
      const int l   = idx / (16 * KROW);
      const int rem = idx - l * (16 * KROW);
      const int m   = rem / KROW;
      const int k   = rem - m * KROW;
      const int tap = k >> 4, c = k & 15;
      float v = 0.0f;
      if (m < 12 && tap < 9 && c < Cin[l])
        v = Wg[l][(m * Cin[l] + c) * 9 + tap];
      wA[0][idx] = (_Float16)v;
    }
  }

#ifdef ASYNC_LDS
#if __has_builtin(__builtin_amdgcn_s_wait_asynccnt)
  __builtin_amdgcn_s_wait_asynccnt(0);
#else
  asm volatile("s_wait_asynccnt 0x0" ::: "memory");
#endif
  __syncthreads();
  // Convert staged f32 tile -> f16 channels 0..3 of S0.
  for (int idx = tid; idx < 22 * 22; idx += 256) {
    const int j  = idx / 22;
    const int i  = idx - j * 22;
    const int iy = gy0 + j - 3;
    const int ix = gx0 + i - 3;
    if (iy >= 0 && iy < IMG && ix >= 0 && ix < IMG) {
      _Float16* dst = &S[0][((j + 1) * AS + (i + 1)) * CH];
      dst[0] = (_Float16)mstage[idx];
      dst[1] = (_Float16)mstage[484 + idx];
      dst[2] = (_Float16)mstage[2 * 484 + idx];
      dst[3] = (_Float16)mstage[3 * 484 + idx];
    }
  }
#else
  __syncthreads();
  // Fallback: direct global->VGPR->LDS staging of the mosaic tile.
  for (int idx = tid; idx < 22 * 22; idx += 256) {
    const int j  = idx / 22;
    const int i  = idx - j * 22;
    const int iy = gy0 + j - 3;
    const int ix = gx0 + i - 3;
    if (iy >= 0 && iy < IMG && ix >= 0 && ix < IMG) {
      _Float16* dst = &S[0][((j + 1) * AS + (i + 1)) * CH];
      const float* mp = mosaic + ((size_t)b * 4) * PLANE + (size_t)iy * IMG + ix;
      dst[0] = (_Float16)mp[0];
      dst[1] = (_Float16)mp[PLANE];
      dst[2] = (_Float16)mp[2 * PLANE];
      dst[3] = (_Float16)mp[3 * PLANE];
    }
  }
#endif
  __syncthreads();

  // f-stack: S0 -> S1 -> S2 -> S1   (layer 3 only needs the 16x16 center)
  conv3x3_wmma<false>(S[0], S[1], wA[0], zeroChunk); __syncthreads();
  conv3x3_wmma<false>(S[1], S[2], wA[1], zeroChunk); __syncthreads();
  conv3x3_wmma<true >(S[2], S[1], wA[2], zeroChunk); __syncthreads();

  // Save f-stack center 16x16x16 before S1 gets reused.
  for (int idx = tid; idx < TW * TW; idx += 256) {
    const int oy = idx >> 4, ox = idx & 15;
    const uint4* s = ldsq(&S[1][((oy + 4) * AS + (ox + 4)) * CH]);
    uint4* d = reinterpret_cast<uint4*>(&interpsF[idx * CH]);
    d[0] = s[0]; d[1] = s[1];
  }
  __syncthreads();

  // w-stack: S0 -> S2 -> S1 -> S2   (mosaic in S0 untouched by f-stack)
  conv3x3_wmma<false>(S[0], S[2], wA[3], zeroChunk); __syncthreads();
  conv3x3_wmma<false>(S[2], S[1], wA[4], zeroChunk); __syncthreads();
  conv3x3_wmma<true >(S[1], S[2], wA[5], zeroChunk); __syncthreads();

  // Softmax over 12 channels, weighted sums -> green at R/B sites; emit the
  // G output plane and the low-res green workspace for the chroma kernel.
  {
    const int oy = tid >> 4, ox = tid & 15;
    const int y = gy0 + oy, x = gx0 + ox;
    const _Float16* wp = &S[2][((oy + 4) * AS + (ox + 4)) * CH];
    const _Float16* fp = &interpsF[tid * CH];
    float wv[12], fv[12], mx = -1e30f;
    #pragma unroll
    for (int i = 0; i < 12; ++i) {
      wv[i] = (float)wp[i]; fv[i] = (float)fp[i];
      mx = fmaxf(mx, wv[i]);
    }
    float den = 0.0f, num0 = 0.0f, num1 = 0.0f;
    #pragma unroll
    for (int i = 0; i < 12; ++i) {
      const float e = __expf(wv[i] - mx);
      den += e;
      if (i < 6) num0 += fv[i] * e; else num1 += fv[i] * e;
    }
    const float g0 = num0 / den;
    const float g1 = num1 / den;

    const size_t moff = ((size_t)b * 4) * PLANE + (size_t)y * IMG + x;
    const float m0 = mosaic[moff];
    const float m3 = mosaic[moff + 3 * PLANE];

    gws[((size_t)b * 2 + 0) * PLANE + (size_t)y * IMG + x] = g0;
    gws[((size_t)b * 2 + 1) * PLANE + (size_t)y * IMG + x] = g1;

    float* og = out + ((size_t)b * 3 + 1) * OPLANE;
    og[(size_t)(2 * y)     * 1024 + 2 * x]     = m0;
    og[(size_t)(2 * y)     * 1024 + 2 * x + 1] = g0;
    og[(size_t)(2 * y + 1) * 1024 + 2 * x]     = g1;
    og[(size_t)(2 * y + 1) * 1024 + 2 * x + 1] = m3;
  }
}

// ---------------------------------------------------------------------------
// Kernel 2: chroma = flat_mosaic - flat_green is nonzero only at the two
// non-G Bayer sites: c01[y][x] at (2y,2x+1) = m1-g0, c10[y][x] at (2y+1,2x)
// = m2-g1. 5x5 convs over that sparse plane reduce to ~12 LDS MACs each.
// ---------------------------------------------------------------------------
__device__ __forceinline__
float conv5(const float* __restrict__ w25, const float* __restrict__ c01,
            const float* __restrict__ c10, int j, int i, int sy, int sx)
{
  float s = 0.0f;
  #pragma unroll
  for (int ky = 0; ky < 5; ++ky) {
    const int dlt = sy + ky - 2;
    #pragma unroll
    for (int kx = 0; kx < 5; ++kx) {
      const int eps = sx + kx - 2;
      if ((dlt & 1) == 0 && (eps & 1) != 0)        // (even,odd) -> c01
        s += w25[ky * 5 + kx] *
             c01[(j + (dlt >> 1) + 2) * 20 + (i + ((eps - 1) >> 1) + 2)];
      else if ((dlt & 1) != 0 && (eps & 1) == 0)   // (odd,even) -> c10
        s += w25[ky * 5 + kx] *
             c10[(j + ((dlt - 1) >> 1) + 2) * 20 + (i + (eps >> 1) + 2)];
    }
  }
  return s;
}

__global__ __launch_bounds__(256)
void demosaic_chroma_kernel(const float* __restrict__ mosaic,
                            const float* __restrict__ chw,
                            const float* __restrict__ cvw,
                            const float* __restrict__ cqw,
                            const float* __restrict__ gws,
                            float* __restrict__ out)
{
  __shared__ float c01[20 * 20];
  __shared__ float c10[20 * 20];
  __shared__ float cw[3][25];

  const int tid = threadIdx.x;
  const int b   = blockIdx.z;
  const int gx0 = blockIdx.x * TW;
  const int gy0 = blockIdx.y * TW;

  for (int idx = tid; idx < 400; idx += 256) {
    const int jj = idx / 20;
    const int ii = idx - jj * 20;
    const int y = gy0 + jj - 2;
    const int x = gx0 + ii - 2;
    float a = 0.0f, c = 0.0f;
    if (y >= 0 && y < IMG && x >= 0 && x < IMG) {
      const size_t off = (size_t)y * IMG + x;
      a = mosaic[((size_t)b * 4 + 1) * PLANE + off] - gws[((size_t)b * 2 + 0) * PLANE + off];
      c = mosaic[((size_t)b * 4 + 2) * PLANE + off] - gws[((size_t)b * 2 + 1) * PLANE + off];
    }
    c01[idx] = a;
    c10[idx] = c;
  }
  if (tid < 75) {
    const float* w3[3] = {chw, cvw, cqw};
    cw[tid / 25][tid % 25] = w3[tid / 25][tid % 25];
  }
  __syncthreads();

  const int j = tid >> 4, i = tid & 15;
  const int y = gy0 + j, x = gx0 + i;

  const size_t moff = ((size_t)b * 4) * PLANE + (size_t)y * IMG + x;
  const float m0 = mosaic[moff];
  const float m1 = mosaic[moff + PLANE];
  const float m2 = mosaic[moff + 2 * PLANE];
  const float m3 = mosaic[moff + 3 * PLANE];
  const float g0 = gws[((size_t)b * 2 + 0) * PLANE + (size_t)y * IMG + x];
  const float g1 = gws[((size_t)b * 2 + 1) * PLANE + (size_t)y * IMG + x];

  const float ch00 = conv5(cw[0], c01, c10, j, i, 0, 0) + m0;
  const float ch11 = conv5(cw[0], c01, c10, j, i, 1, 1) + m3;
  const float cv00 = conv5(cw[1], c01, c10, j, i, 0, 0) + m0;
  const float cv11 = conv5(cw[1], c01, c10, j, i, 1, 1) + m3;
  const float cq10 = conv5(cw[2], c01, c10, j, i, 1, 0) + g1;
  const float cq01 = conv5(cw[2], c01, c10, j, i, 0, 1) + g0;

  float* orp = out + ((size_t)b * 3 + 0) * OPLANE;
  float* obp = out + ((size_t)b * 3 + 2) * OPLANE;
  orp[(size_t)(2 * y)     * 1024 + 2 * x]     = ch00;
  orp[(size_t)(2 * y)     * 1024 + 2 * x + 1] = m1;
  orp[(size_t)(2 * y + 1) * 1024 + 2 * x]     = cq10;
  orp[(size_t)(2 * y + 1) * 1024 + 2 * x + 1] = cv11;
  obp[(size_t)(2 * y)     * 1024 + 2 * x]     = cv00;
  obp[(size_t)(2 * y)     * 1024 + 2 * x + 1] = cq01;
  obp[(size_t)(2 * y + 1) * 1024 + 2 * x]     = m2;
  obp[(size_t)(2 * y + 1) * 1024 + 2 * x + 1] = ch11;
}

extern "C" void kernel_launch(void* const* d_in, const int* in_sizes, int n_in,
                              void* d_out, int out_size, void* d_ws, size_t ws_size,
                              hipStream_t stream) {
  const float* mosaic = (const float*)d_in[0];
  const float* fw0 = (const float*)d_in[1];
  const float* fw1 = (const float*)d_in[2];
  const float* fw2 = (const float*)d_in[3];
  const float* ww0 = (const float*)d_in[4];
  const float* ww1 = (const float*)d_in[5];
  const float* ww2 = (const float*)d_in[6];
  const float* chw = (const float*)d_in[7];
  const float* cvw = (const float*)d_in[8];
  const float* cqw = (const float*)d_in[9];
  float* out = (float*)d_out;
  float* gws = (float*)d_ws;   // needs 8*2*512*512*4 = 16 MiB of scratch

  dim3 grid(IMG / TW, IMG / TW, 8);   // 32 x 32 x 8 tiles
  dim3 blk(256);
  demosaic_stacks_kernel<<<grid, blk, 0, stream>>>(mosaic, fw0, fw1, fw2,
                                                   ww0, ww1, ww2, out, gws);
  demosaic_chroma_kernel<<<grid, blk, 0, stream>>>(mosaic, chw, cvw, cqw, gws, out);
}
